// Attention_62715112456252
// MI455X (gfx1250) — compile-verified, matrix operand order checked by reading, not run on hardware
//
#include <hip/hip_runtime.h>
#include <hip/hip_bf16.h>

// ---------------------------------------------------------------------------
// Problem constants (must match reference)
// ---------------------------------------------------------------------------
#define BATCH      4
#define TOKENS     2048          // NUM_PATCHES * NUM_FRAMES
#define DIM        512
#define HEADS      8
#define DHEAD      64
#define INNER      512           // HEADS * DHEAD
#define NQKV       1536          // 3 * INNER
#define MROWS      (BATCH * TOKENS)   // 8192
#define FRAME      64            // NUM_PATCHES (tokens per frame)
#define NFRAMES    32
#define EPS        1e-5f

typedef __attribute__((ext_vector_type(16))) _Float16 v16h;
typedef __attribute__((ext_vector_type(8)))  float    v8f;
typedef __attribute__((ext_vector_type(4)))  int      v4i;

#define WMMA(A, B, C) \
  __builtin_amdgcn_wmma_f32_16x16x32_f16(false, (A), false, (B), (short)0, (C), false, false)

// ---------------------------------------------------------------------------
// Async Global->LDS copy (gfx1250 GLOBAL_LOAD_ASYNC_TO_LDS_B128, ASYNCcnt).
// Builtin wants (global int4*, shared int4*, imm offset, imm cpol).
// Guarded: falls back to global->VGPR->ds_store if the builtin is absent.
// ---------------------------------------------------------------------------
#if defined(__AMDGCN__) && __has_builtin(__builtin_amdgcn_global_load_async_to_lds_b128)
#define ASYNC_CP 1
#define GAS __attribute__((address_space(1)))
#define LAS __attribute__((address_space(3)))
#else
#define ASYNC_CP 0
#endif

__device__ __forceinline__ void cp16_async(const _Float16* g, _Float16* l) {
#if ASYNC_CP
  __builtin_amdgcn_global_load_async_to_lds_b128(
      (GAS v4i*)(void*)g, (LAS v4i*)l, 0, 0);
#else
  *(float4*)l = *(const float4*)g;
#endif
}

__device__ __forceinline__ void cp_wait() {
#if ASYNC_CP
#if __has_builtin(__builtin_amdgcn_s_wait_asynccnt)
  __builtin_amdgcn_s_wait_asynccnt(0);
#else
  asm volatile("s_wait_asynccnt 0" ::: "memory");
#endif
#endif
}

// ---------------------------------------------------------------------------
// Fragment gather: 16-bit A-matrix 16x32 layout (CDNA5 ISA 7.12.2).
// lane L: index0 = L&15 (M for A, N for B); VGPR r holds K pair
//   k = 16*(r>=4) + 8*(L>=16) + 2*(r&3) + {0,1}
// Source tile is row-major [idx][k] with row stride `ld` (halfs).
// Works for both A tiles and (N-major, i.e. pre-transposed) B tiles.
// ---------------------------------------------------------------------------
__device__ __forceinline__ v16h ldfrag(const _Float16* __restrict__ p, int ld) {
  const int lane = threadIdx.x & 31;
  const int i0 = lane & 15;
  const int h8 = (lane & 16) >> 1;   // 0 or 8
  v16h f;
#pragma unroll
  for (int r = 0; r < 8; ++r) {
    const int k = ((r & 4) << 2) | h8 | ((r & 3) << 1);
    f[2 * r]     = p[i0 * ld + k];
    f[2 * r + 1] = p[i0 * ld + k + 1];
  }
  return f;
}

// ---------------------------------------------------------------------------
// Kernel 1: LayerNorm (f32 in) -> f16 normalized activations
// ---------------------------------------------------------------------------
__global__ __launch_bounds__(256) void ln_kernel(
    const float* __restrict__ x, const float* __restrict__ g,
    const float* __restrict__ b, _Float16* __restrict__ xn) {
  __shared__ float rs[8], rs2[8];
  const int row = blockIdx.x;
  const float* xr = x + (size_t)row * DIM;
  float s = 0.f, s2 = 0.f;
  for (int i = threadIdx.x; i < DIM; i += 256) {
    float v = xr[i];
    s += v; s2 += v * v;
  }
#pragma unroll
  for (int m = 1; m < 32; m <<= 1) {
    s  += __shfl_xor(s, m);
    s2 += __shfl_xor(s2, m);
  }
  const int wid = threadIdx.x >> 5;
  if ((threadIdx.x & 31) == 0) { rs[wid] = s; rs2[wid] = s2; }
  __syncthreads();
  float sum = 0.f, sum2 = 0.f;
#pragma unroll
  for (int w = 0; w < 8; ++w) { sum += rs[w]; sum2 += rs2[w]; }
  const float mu  = sum * (1.0f / DIM);
  const float var = sum2 * (1.0f / DIM) - mu * mu;
  const float inv = rsqrtf(var + EPS);
  for (int i = threadIdx.x; i < DIM; i += 256)
    xn[(size_t)row * DIM + i] = (_Float16)((xr[i] - mu) * inv * g[i] + b[i]);
}

// ---------------------------------------------------------------------------
// Kernel 2: weight downcast f32 -> f16
// ---------------------------------------------------------------------------
__global__ __launch_bounds__(256) void cvt_kernel(
    const float* __restrict__ wqkv, const float* __restrict__ wout,
    _Float16* __restrict__ wqkvh, _Float16* __restrict__ wouth) {
  const int i = blockIdx.x * 256 + threadIdx.x;
  if (i < DIM * NQKV) wqkvh[i] = (_Float16)wqkv[i];
  if (i < DIM * INNER) wouth[i] = (_Float16)wout[i];
}

// ---------------------------------------------------------------------------
// GEMM tiles: block 256 threads = 8 waves (4M x 2N), tile 128x128, k-step 64.
// A tile copied via async global->LDS; B tile transposed at store ([n][k]).
// ---------------------------------------------------------------------------
#define GLDA 72   // padded LDS row stride (halfs), 16B aligned (72 = 9*8)

// ---------------------------------------------------------------------------
// Kernel 3: QKV GEMM (M=8192, K=512, N=1536) -> head-major Q/K/V f16
// ---------------------------------------------------------------------------
__global__ __launch_bounds__(256) void qkv_gemm(
    const _Float16* __restrict__ xn, const _Float16* __restrict__ w,
    _Float16* __restrict__ Q, _Float16* __restrict__ K, _Float16* __restrict__ V) {
  __shared__ _Float16 sA[128 * GLDA];   // [m][k]
  __shared__ _Float16 sBt[128 * GLDA];  // [n][k] (transposed at store)
  const int tid = threadIdx.x;
  const int m0 = blockIdx.y * 128;
  const int n0 = blockIdx.x * 128;
  const int wave = tid >> 5;
  const int rowA = (wave & 3) * 32;
  const int colB = (wave >> 2) * 64;
  v8f acc[2][4] = {};

  for (int k0 = 0; k0 < DIM; k0 += 64) {
    __syncthreads();
    // A tile: 128 rows x 64 halfs -> 1024 slots of 8 halfs (async to LDS)
    for (int s = tid; s < 1024; s += 256) {
      const int r = s >> 3, kg = (s & 7) << 3;
      cp16_async(xn + (size_t)(m0 + r) * DIM + k0 + kg, &sA[r * GLDA + kg]);
    }
    // B tile (transpose at store): 64 k-rows x 128 n
    for (int s = tid; s < 1024; s += 256) {
      const int kk = s >> 4, ng = (s & 15) << 3;
      _Float16 tmp[8];
      *(float4*)tmp = *(const float4*)(w + (size_t)(k0 + kk) * NQKV + n0 + ng);
#pragma unroll
      for (int j = 0; j < 8; ++j) sBt[(ng + j) * GLDA + kk] = tmp[j];
    }
    cp_wait();
    __syncthreads();
#pragma unroll
    for (int kc = 0; kc < 2; ++kc) {
      const v16h a0 = ldfrag(&sA[(rowA + 0) * GLDA + kc * 32], GLDA);
      const v16h a1 = ldfrag(&sA[(rowA + 16) * GLDA + kc * 32], GLDA);
#pragma unroll
      for (int ni = 0; ni < 4; ++ni) {
        const v16h bf = ldfrag(&sBt[(colB + ni * 16) * GLDA + kc * 32], GLDA);
        acc[0][ni] = WMMA(a0, bf, acc[0][ni]);
        acc[1][ni] = WMMA(a1, bf, acc[1][ni]);
      }
    }
  }

  const int lane = tid & 31, l15 = lane & 15, hf = lane >> 4;
#pragma unroll
  for (int mi = 0; mi < 2; ++mi)
#pragma unroll
    for (int ni = 0; ni < 4; ++ni) {
      const int n = n0 + colB + ni * 16 + l15;
      const int which = n / INNER, inner = n % INNER;
      const int h = inner >> 6, d = inner & 63;
      _Float16* dst = (which == 0) ? Q : (which == 1) ? K : V;
#pragma unroll
      for (int r = 0; r < 8; ++r) {
        const int m = m0 + rowA + mi * 16 + r + 8 * hf;
        const int b = m >> 11, t = m & 2047;
        dst[(((size_t)(b * HEADS + h)) * TOKENS + t) * DHEAD + d] =
            (_Float16)acc[mi][ni][r];
      }
    }
}

// ---------------------------------------------------------------------------
// Kernel 4: flash attention with block-causal (frame) mask.
// Grid: (frame-block qb in [0,32), b*h in [0,32)). Block: 128 threads = 4 waves.
// Each wave owns 16 query rows; key loop runs over (qb+1) chunks of 64 keys —
// frame-causality is exactly tile-aligned, so no element masking is needed.
// ---------------------------------------------------------------------------
#define LDK 72    // padded LDS stride (halfs), 16B aligned
__global__ __launch_bounds__(128) void attn_kernel(
    const _Float16* __restrict__ Q, const _Float16* __restrict__ K,
    const _Float16* __restrict__ V, _Float16* __restrict__ O) {
  __shared__ _Float16 sK[64 * LDK];      // [key][d] row-major
  __shared__ _Float16 sVt[64 * LDK];     // [d][key] transposed
  __shared__ _Float16 sP[4][16 * 64];    // per-wave P scratch [row][key]
  const int tid = threadIdx.x;
  const int wave = tid >> 5;
  const int lane = tid & 31, l15 = lane & 15, hf = lane >> 4;
  const int qb = blockIdx.x;   // frame block (== frame index of these queries)
  const int bh = blockIdx.y;   // b*HEADS + h
  const float scale = 0.125f;  // 64^-0.5

  const _Float16* Qg = Q + ((size_t)bh * TOKENS + qb * FRAME + wave * 16) * DHEAD;
  const _Float16* Kg = K + (size_t)bh * TOKENS * DHEAD;
  const _Float16* Vg = V + (size_t)bh * TOKENS * DHEAD;

  const v16h aQ0 = ldfrag(Qg, DHEAD);
  const v16h aQ1 = ldfrag(Qg + 32, DHEAD);

  float mrow[8], lrow[8];
  v8f oacc[4] = {};
#pragma unroll
  for (int r = 0; r < 8; ++r) { mrow[r] = -3.0e38f; lrow[r] = 0.f; }

  const int nkc = qb + 1;
  for (int kc = 0; kc < nkc; ++kc) {
    __syncthreads();
    // cooperative K/V chunk load (64 rows x 64 halfs each);
    // K goes async straight to LDS, V is transposed through VGPRs.
    for (int s = tid; s < 512; s += 128) {
      const int r = s >> 3, dg = (s & 7) << 3;
      const size_t gro = (size_t)(kc * FRAME + r) * DHEAD + dg;
      cp16_async(Kg + gro, &sK[r * LDK + dg]);
      _Float16 tmp[8];
      *(float4*)tmp = *(const float4*)(Vg + gro);
#pragma unroll
      for (int j = 0; j < 8; ++j) sVt[(dg + j) * LDK + r] = tmp[j];
    }
    cp_wait();
    __syncthreads();

    // S[16 x 64] = Q Kt  (4 col-subtiles, contraction over d=64 in 2 steps)
    v8f S[4] = {};
#pragma unroll
    for (int s4 = 0; s4 < 4; ++s4) {
      const v16h b0 = ldfrag(&sK[(s4 * 16) * LDK], LDK);
      S[s4] = WMMA(aQ0, b0, S[s4]);
      const v16h b1 = ldfrag(&sK[(s4 * 16) * LDK + 32], LDK);
      S[s4] = WMMA(aQ1, b1, S[s4]);
    }

    // online softmax update (rows live in 16-lane halves of the wave)
    float alpha[8];
#pragma unroll
    for (int r = 0; r < 8; ++r) {
      float v = S[0][r];
      v = fmaxf(v, S[1][r]); v = fmaxf(v, S[2][r]); v = fmaxf(v, S[3][r]);
      v *= scale;
      v = fmaxf(v, __shfl_xor(v, 1));
      v = fmaxf(v, __shfl_xor(v, 2));
      v = fmaxf(v, __shfl_xor(v, 4));
      v = fmaxf(v, __shfl_xor(v, 8));
      const float mn = fmaxf(mrow[r], v);
      alpha[r] = __expf(mrow[r] - mn);
      mrow[r] = mn;
    }
    float rsum[8] = {0.f, 0.f, 0.f, 0.f, 0.f, 0.f, 0.f, 0.f};
    _Float16* pS = &sP[wave][0];
#pragma unroll
    for (int s4 = 0; s4 < 4; ++s4)
#pragma unroll
      for (int r = 0; r < 8; ++r) {
        const float p = __expf(S[s4][r] * scale - mrow[r]);
        rsum[r] += p;
        pS[(r + 8 * hf) * 64 + s4 * 16 + l15] = (_Float16)p;
      }
#pragma unroll
    for (int r = 0; r < 8; ++r) {
      float v = rsum[r];
      v += __shfl_xor(v, 1);
      v += __shfl_xor(v, 2);
      v += __shfl_xor(v, 4);
      v += __shfl_xor(v, 8);
      lrow[r] = lrow[r] * alpha[r] + v;
    }
#pragma unroll
    for (int os = 0; os < 4; ++os)
#pragma unroll
      for (int r = 0; r < 8; ++r) oacc[os][r] *= alpha[r];

    // O += P V  (P via per-wave LDS round-trip to A layout; same-wave DS is in order)
    const v16h aP0 = ldfrag(pS, 64);
    const v16h aP1 = ldfrag(pS + 32, 64);
#pragma unroll
    for (int os = 0; os < 4; ++os) {
      const v16h bv0 = ldfrag(&sVt[(os * 16) * LDK], LDK);
      oacc[os] = WMMA(aP0, bv0, oacc[os]);
      const v16h bv1 = ldfrag(&sVt[(os * 16) * LDK + 32], LDK);
      oacc[os] = WMMA(aP1, bv1, oacc[os]);
    }
  }

  // normalize + store to [B][T][INNER] f16
  const int b = bh >> 3, h = bh & 7;
#pragma unroll
  for (int os = 0; os < 4; ++os)
#pragma unroll
    for (int r = 0; r < 8; ++r) {
      const int t = qb * FRAME + wave * 16 + r + 8 * hf;
      const int d = os * 16 + l15;
      O[((size_t)(b * TOKENS + t)) * INNER + h * DHEAD + d] =
          (_Float16)(oacc[os][r] / lrow[r]);
    }
}

// ---------------------------------------------------------------------------
// Kernel 5: output projection GEMM (M=8192, K=512, N=512) + bias, f32 out
// ---------------------------------------------------------------------------
__global__ __launch_bounds__(256) void out_gemm(
    const _Float16* __restrict__ a, const _Float16* __restrict__ w,
    const float* __restrict__ bias, float* __restrict__ out) {
  __shared__ _Float16 sA[128 * GLDA];
  __shared__ _Float16 sBt[128 * GLDA];
  const int tid = threadIdx.x;
  const int m0 = blockIdx.y * 128;
  const int n0 = blockIdx.x * 128;
  const int wave = tid >> 5;
  const int rowA = (wave & 3) * 32;
  const int colB = (wave >> 2) * 64;
  v8f acc[2][4] = {};

  for (int k0 = 0; k0 < INNER; k0 += 64) {
    __syncthreads();
    for (int s = tid; s < 1024; s += 256) {
      const int r = s >> 3, kg = (s & 7) << 3;
      cp16_async(a + (size_t)(m0 + r) * INNER + k0 + kg, &sA[r * GLDA + kg]);
    }
    for (int s = tid; s < 1024; s += 256) {
      const int kk = s >> 4, ng = (s & 15) << 3;
      _Float16 tmp[8];
      *(float4*)tmp = *(const float4*)(w + (size_t)(k0 + kk) * DIM + n0 + ng);
#pragma unroll
      for (int j = 0; j < 8; ++j) sBt[(ng + j) * GLDA + kk] = tmp[j];
    }
    cp_wait();
    __syncthreads();
#pragma unroll
    for (int kc = 0; kc < 2; ++kc) {
      const v16h a0 = ldfrag(&sA[(rowA + 0) * GLDA + kc * 32], GLDA);
      const v16h a1 = ldfrag(&sA[(rowA + 16) * GLDA + kc * 32], GLDA);
#pragma unroll
      for (int ni = 0; ni < 4; ++ni) {
        const v16h bf = ldfrag(&sBt[(colB + ni * 16) * GLDA + kc * 32], GLDA);
        acc[0][ni] = WMMA(a0, bf, acc[0][ni]);
        acc[1][ni] = WMMA(a1, bf, acc[1][ni]);
      }
    }
  }

  const int lane = tid & 31, l15 = lane & 15, hf = lane >> 4;
#pragma unroll
  for (int mi = 0; mi < 2; ++mi)
#pragma unroll
    for (int ni = 0; ni < 4; ++ni) {
      const int n = n0 + colB + ni * 16 + l15;
      const float bv = bias[n];
#pragma unroll
      for (int r = 0; r < 8; ++r) {
        const int m = m0 + rowA + mi * 16 + r + 8 * hf;
        out[(size_t)m * DIM + n] = acc[mi][ni][r] + bv;
      }
    }
}

// ---------------------------------------------------------------------------
// Host launcher
// ---------------------------------------------------------------------------
extern "C" void kernel_launch(void* const* d_in, const int* in_sizes, int n_in,
                              void* d_out, int out_size, void* d_ws, size_t ws_size,
                              hipStream_t stream) {
  const float* x     = (const float*)d_in[0];
  const float* ln_g  = (const float*)d_in[1];
  const float* ln_b  = (const float*)d_in[2];
  const float* w_qkv = (const float*)d_in[3];
  const float* w_out = (const float*)d_in[4];
  const float* b_out = (const float*)d_in[5];
  float* out = (float*)d_out;

  char* ws = (char*)d_ws;
  _Float16* xn    = (_Float16*)ws;  ws += (size_t)MROWS * DIM * 2;              // 8 MiB
  _Float16* wqkvh = (_Float16*)ws;  ws += (size_t)DIM * NQKV * 2;               // 1.5 MiB
  _Float16* wouth = (_Float16*)ws;  ws += (size_t)INNER * DIM * 2;              // 0.5 MiB
  _Float16* Qb    = (_Float16*)ws;  ws += (size_t)BATCH * HEADS * TOKENS * DHEAD * 2; // 8 MiB
  _Float16* Kb    = (_Float16*)ws;  ws += (size_t)BATCH * HEADS * TOKENS * DHEAD * 2; // 8 MiB
  _Float16* Vb    = (_Float16*)ws;  ws += (size_t)BATCH * HEADS * TOKENS * DHEAD * 2; // 8 MiB
  _Float16* attn  = xn;  // xn is dead after qkv_gemm; reuse (stream-ordered)

  ln_kernel<<<MROWS, 256, 0, stream>>>(x, ln_g, ln_b, xn);
  cvt_kernel<<<(DIM * NQKV + 255) / 256, 256, 0, stream>>>(w_qkv, w_out, wqkvh, wouth);
  qkv_gemm<<<dim3(NQKV / 128, MROWS / 128), 256, 0, stream>>>(xn, wqkvh, Qb, Kb, Vb);
  attn_kernel<<<dim3(NFRAMES, BATCH * HEADS), 128, 0, stream>>>(Qb, Kb, Vb, attn);
  out_gemm<<<dim3(DIM / 128, MROWS / 128), 256, 0, stream>>>(attn, wouth, b_out, out);
}
